// ExplicitSVDBlock_37812892074280
// MI455X (gfx1250) — matrix operand order repaired and checked by reading, not compiled, not run on hardware
//
#include <hip/hip_runtime.h>
#include <math.h>

// ---------------------------------------------------------------------------
// MI455X (gfx1250) implementation: bf16 WMMA (v_wmma_f32_16x16x32_bf16),
// wave32 fragment layouts per CDNA5 ISA 7.12.2, flash-style attention to
// avoid the 1GB score tensor, async global->LDS staging (ASYNCcnt) for GEMM
// tiles, all matmuls on the matrix pipeline.
// ---------------------------------------------------------------------------

typedef __attribute__((ext_vector_type(16))) __bf16 v16bf;
typedef __attribute__((ext_vector_type(8)))  float  v8f;

#define DEV __device__ __forceinline__

DEV unsigned short f2b(float f) {
  unsigned int u = __float_as_uint(f);
  unsigned int r = (u + 0x7FFFu + ((u >> 16) & 1u)) >> 16;  // RNE
  return (unsigned short)r;
}
DEV float b2f(unsigned short h) { return __uint_as_float(((unsigned int)h) << 16); }

union Frag {
  uint4 q[2];
  v16bf v;
  unsigned short s[16];
};

// ---- async global->LDS copy, 16 bytes per lane (GLOBAL_LOAD_ASYNC_TO_LDS) --
// the LDS address is taken from the VDST VGPR; generic LDS pointers keep the wave-relative
// LDS byte offset in their low 32 bits (flat aperture, ISA 10.2), so a
// truncating cast produces the VDST operand.
#if defined(__gfx1250__)
#define USE_ASYNC_LDS 1
#else
#define USE_ASYNC_LDS 0
#endif

DEV void g2l_b128(const unsigned short* g, unsigned short* l) {
#if USE_ASYNC_LDS
  unsigned int lds_off = (unsigned int)(unsigned long long)l;
  asm volatile("global_load_async_to_lds_b128 %0, %1, off"
               :: "v"(lds_off), "v"(g)
               : "memory");
#else
  *(uint4*)l = *(const uint4*)g;
#endif
}
DEV void g2l_wait() {
#if USE_ASYNC_LDS
  asm volatile("s_wait_asynccnt 0x0" ::: "memory");
#endif
}

// --------------------------- f32 -> bf16 convert ---------------------------
__global__ void __launch_bounds__(256) cvt_kernel(
    const float* __restrict__ in, unsigned short* __restrict__ out, int n) {
  int i = blockIdx.x * 256 + threadIdx.x;
  if (i < n) out[i] = f2b(in[i]);
}

// ------------------------------- LayerNorm ---------------------------------
// block = 256 threads, one row of length Dd per block; bf16 output.
__global__ void __launch_bounds__(256) ln_kernel(
    const float* __restrict__ x, const float* __restrict__ w,
    const float* __restrict__ b, unsigned short* __restrict__ out, int Dd) {
  __shared__ float red[2][8];
  int row = blockIdx.x;
  int t = threadIdx.x;
  const float* xr = x + (size_t)row * Dd;
  float s = 0.f, ss = 0.f;
  for (int i = t; i < Dd; i += 256) { float v = xr[i]; s += v; ss += v * v; }
  for (int o = 16; o > 0; o >>= 1) {
    s  += __shfl_down(s,  o, 32);
    ss += __shfl_down(ss, o, 32);
  }
  int wid = t >> 5, lane = t & 31;
  if (lane == 0) { red[0][wid] = s; red[1][wid] = ss; }
  __syncthreads();
  if (t == 0) {
    float a = 0.f, c = 0.f;
    for (int i = 0; i < 8; i++) { a += red[0][i]; c += red[1][i]; }
    red[0][0] = a; red[1][0] = c;
  }
  __syncthreads();
  float mu  = red[0][0] / (float)Dd;
  float var = red[1][0] / (float)Dd - mu * mu;
  float rstd = rsqrtf(var + 1e-5f);
  unsigned short* orow = out + (size_t)row * Dd;
  for (int i = t; i < Dd; i += 256)
    orow[i] = f2b((xr[i] - mu) * rstd * w[i] + b[i]);
}

// --------------------------- RoPE / head transpose -------------------------
// in:  [B, M, H*64] bf16 ; out: [B, H, M, 64] bf16 (rope applied if doRope)
__global__ void __launch_bounds__(256) rope_kernel(
    const unsigned short* __restrict__ in, unsigned short* __restrict__ out,
    int doRope) {
  int idx = blockIdx.x * 256 + threadIdx.x;   // B*M*H*32 threads
  int i = idx & 31;
  int h = (idx >> 5) & 15;
  int m = (idx >> 9) & 2047;
  int b = idx >> 20;
  const unsigned short* ip = in + (((size_t)b * 2048 + m) * 1024 + h * 64);
  unsigned short* op = out + ((((size_t)b * 16 + h) * 2048 + m) * 64);
  float x1 = b2f(ip[i]);
  float x2 = b2f(ip[i + 32]);
  if (doRope) {
    float ang = (float)m * __powf(10000.0f, -(float)i / 32.0f);
    float sn, cs;
    __sincosf(ang, &sn, &cs);
    op[i]      = f2b(x1 * cs - x2 * sn);
    op[i + 32] = f2b(x2 * cs + x1 * sn);
  } else {
    op[i]      = ip[i];
    op[i + 32] = ip[i + 32];
  }
}

// ------------------------------ WMMA GEMM ----------------------------------
// C[M,N] = A[M,K] * B + (bias) + (res).  A bf16 row-major (lda=K).
// BT=0: B is [K,N] bf16 (ldb=N).  BT=1: B is [N,K] bf16, op uses B^T.
// OUTB=1: bf16 out to Cb.  OUTB=0: f32 out to Cf.
// Block: 256 threads (8 waves, 4x2), tile 128x64, K step 32.
// A tile (and B^T tile) staged with GLOBAL_LOAD_ASYNC_TO_LDS_B128.
template <int BT, int OUTB, int BIAS, int RES>
__global__ void __launch_bounds__(256) gemm_kernel(
    const unsigned short* __restrict__ A,
    const unsigned short* __restrict__ Bm,
    const float* __restrict__ bias,
    const float* __restrict__ res,
    unsigned short* __restrict__ Cb,
    float* __restrict__ Cf,
    int Mdim, int Ndim, int Kdim) {
  __shared__ unsigned short sA[128][40];  // +8 pad, 16B aligned rows
  __shared__ unsigned short sB[64][40];   // stored as Bt[n][k]
  int t = threadIdx.x;
  int wid = t >> 5, lane = t & 31, half = (lane >> 4) & 1, lr = lane & 15;
  int wm = wid >> 1, wn = wid & 1;
  int m0 = blockIdx.y * 128;
  int n0 = blockIdx.x * 64;
  v8f acc[2][2] = {};

  for (int k0 = 0; k0 < Kdim; k0 += 32) {
    // ---- stage A tile 128x32 (512 x 8-elt chunks; 2 per thread, async) ----
    {
      int c0 = t;
      int row = c0 >> 2, kp = (c0 & 3) << 3;
      g2l_b128(A + (size_t)(m0 + row) * Kdim + k0 + kp, &sA[row][kp]);
      int c1 = t + 256;
      row = c1 >> 2; kp = (c1 & 3) << 3;
      g2l_b128(A + (size_t)(m0 + row) * Kdim + k0 + kp, &sA[row][kp]);
    }
    // ---- stage B tile as Bt[n][k] 64x32 ----
    if (BT) {
      int row = t >> 2, kp = (t & 3) << 3;
      g2l_b128(Bm + (size_t)(n0 + row) * Kdim + k0 + kp, &sB[row][kp]);
    } else {
      int k = t & 31, np = (t >> 5) << 3;
      uint4 v = *(const uint4*)(Bm + (size_t)(k0 + k) * Ndim + n0 + np);
      const unsigned short* sv = (const unsigned short*)&v;
#pragma unroll
      for (int j = 0; j < 8; j++) sB[np + j][k] = sv[j];
    }
    if (k0 + 32 < Kdim) {  // emits global_prefetch_b8
      __builtin_prefetch(A + (size_t)(m0 + (t >> 2)) * Kdim + k0 + 32 + ((t & 3) << 3), 0, 1);
    }
    g2l_wait();
    __syncthreads();

    // ---- fragments ----
    Frag a[2], bf[2];
#pragma unroll
    for (int mt = 0; mt < 2; mt++) {
      int ar = wm * 32 + mt * 16 + lr;  // A row = lane%16
      a[mt].q[0] = *(const uint4*)(&sA[ar][half * 8]);        // K 0..7 / 8..15
      a[mt].q[1] = *(const uint4*)(&sA[ar][16 + half * 8]);   // K 16..23 / 24..31
    }
#pragma unroll
    for (int nt = 0; nt < 2; nt++) {
      int bc = wn * 32 + nt * 16 + lr;  // B col = lane%16
      bf[nt].q[0] = *(const uint4*)(&sB[bc][half * 8]);       // K 0..7 / 8..15
      bf[nt].q[1] = *(const uint4*)(&sB[bc][16 + half * 8]);  // K 16..23 / 24..31
    }
#pragma unroll
    for (int mt = 0; mt < 2; mt++)
#pragma unroll
      for (int nt = 0; nt < 2; nt++)
        acc[mt][nt] = __builtin_amdgcn_wmma_f32_16x16x32_bf16(
            false, a[mt].v, false, bf[nt].v, (short)0, acc[mt][nt], false, false);
    __syncthreads();
  }

  // ---- epilogue ----
#pragma unroll
  for (int mt = 0; mt < 2; mt++) {
#pragma unroll
    for (int nt = 0; nt < 2; nt++) {
      int col = n0 + wn * 32 + nt * 16 + lr;
      float bv = BIAS ? bias[col] : 0.0f;
#pragma unroll
      for (int r = 0; r < 8; r++) {
        int row = m0 + wm * 32 + mt * 16 + r + half * 8;
        size_t idx = (size_t)row * Ndim + col;
        float v = acc[mt][nt][r] + bv;
        if (RES) v += res[idx];
        if (OUTB) Cb[idx] = f2b(v);
        else      Cf[idx] = v;
      }
    }
  }
}

// --------------------------- Flash attention -------------------------------
// Q/K/V: [B, H, M, 64] bf16.  Out: attn_out [B, M, H*64] bf16.
// Block: 128 threads (4 waves); each wave owns 16 query rows; KV blocks of 64.
__global__ void __launch_bounds__(128) flash_kernel(
    const unsigned short* __restrict__ Qr,
    const unsigned short* __restrict__ Kr,
    const unsigned short* __restrict__ Vr,
    const int* __restrict__ mask,
    unsigned short* __restrict__ Ob) {
  __shared__ unsigned short sV[64][72];      // V^T tile: sV[d][key]
  __shared__ unsigned short sP[4][16][72];   // per-wave P (C-layout -> A-layout)
  const int M_ = 2048, H_ = 16;
  int b = blockIdx.z, h = blockIdx.y, q0 = blockIdx.x * 64;
  int t = threadIdx.x, wid = t >> 5, lane = t & 31;
  int half = (lane >> 4) & 1, lr = lane & 15;
  const size_t head = ((size_t)b * H_ + h) * M_;

  // Q A-fragments live in registers for the whole KV sweep
  Frag aQ[2];
  {
    const unsigned short* qrow = Qr + (head + q0 + wid * 16 + lr) * 64;
#pragma unroll
    for (int kk = 0; kk < 2; kk++) {
      aQ[kk].q[0] = *(const uint4*)(qrow + kk * 32 + half * 8);
      aQ[kk].q[1] = *(const uint4*)(qrow + kk * 32 + 16 + half * 8);
    }
  }

  float mrun[8], lrun[8];
  v8f accO[4] = {};
#pragma unroll
  for (int r = 0; r < 8; r++) { mrun[r] = -1e30f; lrun[r] = 0.f; }
  const float scale = 0.125f;  // 1/sqrt(64)

  for (int j = 0; j < 32; j++) {
    int kv0 = j * 64;
    __syncthreads();  // previous iteration's sV/sP reads complete
    // stage V^T (64x64), 4 chunks per thread
    for (int c = t; c < 512; c += 128) {
      int key = c >> 3, dp = (c & 7) << 3;
      uint4 v = *(const uint4*)(Vr + (head + kv0 + key) * 64 + dp);
      const unsigned short* sv = (const unsigned short*)&v;
#pragma unroll
      for (int jj = 0; jj < 8; jj++) sV[dp + jj][key] = sv[jj];
    }
    __syncthreads();

    // S = Q K^T : K fragments load straight from global (d-contiguous)
    v8f sAcc[4];
#pragma unroll
    for (int nt = 0; nt < 4; nt++) {
      const unsigned short* krow = Kr + (head + kv0 + nt * 16 + lr) * 64;
      Frag bk0, bk1;
      bk0.q[0] = *(const uint4*)(krow + half * 8);
      bk0.q[1] = *(const uint4*)(krow + 16 + half * 8);
      bk1.q[0] = *(const uint4*)(krow + 32 + half * 8);
      bk1.q[1] = *(const uint4*)(krow + 32 + 16 + half * 8);
      v8f z = {};
      z = __builtin_amdgcn_wmma_f32_16x16x32_bf16(false, aQ[0].v, false, bk0.v,
                                                  (short)0, z, false, false);
      z = __builtin_amdgcn_wmma_f32_16x16x32_bf16(false, aQ[1].v, false, bk1.v,
                                                  (short)0, z, false, false);
      sAcc[nt] = z;
    }

    // online softmax (row M = r + 8*half lives across lanes (lane&16)==const)
    int mk[4];
#pragma unroll
    for (int nt = 0; nt < 4; nt++) mk[nt] = mask[b * M_ + kv0 + nt * 16 + lr];
#pragma unroll
    for (int r = 0; r < 8; r++) {
      float mx = -1e30f;
#pragma unroll
      for (int nt = 0; nt < 4; nt++) {
        float v = sAcc[nt][r] * scale;
        if (mk[nt] == 0) v = -1e30f;
        sAcc[nt][r] = v;
        mx = fmaxf(mx, v);
      }
#pragma unroll
      for (int o = 1; o < 16; o <<= 1) mx = fmaxf(mx, __shfl_xor(mx, o, 32));
      float mnew = fmaxf(mrun[r], mx);
      float alpha = __expf(mrun[r] - mnew);
      float sum = 0.f;
#pragma unroll
      for (int nt = 0; nt < 4; nt++) {
        float p = __expf(sAcc[nt][r] - mnew);
        sAcc[nt][r] = p;
        sum += p;
      }
#pragma unroll
      for (int o = 1; o < 16; o <<= 1) sum += __shfl_xor(sum, o, 32);
      lrun[r] = lrun[r] * alpha + sum;
      mrun[r] = mnew;
#pragma unroll
      for (int nt = 0; nt < 4; nt++) {
        accO[nt][r] = accO[nt][r] * alpha;
        sP[wid][r + half * 8][nt * 16 + lr] = f2b(sAcc[nt][r]);
      }
    }
    __syncthreads();

    // O += P V
    Frag aP[2];
#pragma unroll
    for (int kk = 0; kk < 2; kk++) {
      aP[kk].q[0] = *(const uint4*)(&sP[wid][lr][kk * 32 + half * 8]);
      aP[kk].q[1] = *(const uint4*)(&sP[wid][lr][kk * 32 + 16 + half * 8]);
    }
#pragma unroll
    for (int nt = 0; nt < 4; nt++) {
      int d = nt * 16 + lr;
      Frag bv0, bv1;
      bv0.q[0] = *(const uint4*)(&sV[d][half * 8]);
      bv0.q[1] = *(const uint4*)(&sV[d][16 + half * 8]);
      bv1.q[0] = *(const uint4*)(&sV[d][32 + half * 8]);
      bv1.q[1] = *(const uint4*)(&sV[d][32 + 16 + half * 8]);
      accO[nt] = __builtin_amdgcn_wmma_f32_16x16x32_bf16(
          false, aP[0].v, false, bv0.v, (short)0, accO[nt], false, false);
      accO[nt] = __builtin_amdgcn_wmma_f32_16x16x32_bf16(
          false, aP[1].v, false, bv1.v, (short)0, accO[nt], false, false);
    }
  }

  // write attn_out [B, M, H*64]
#pragma unroll
  for (int r = 0; r < 8; r++) {
    float inv = (lrun[r] > 0.f) ? 1.0f / lrun[r] : 0.f;
    int m = q0 + wid * 16 + r + half * 8;
    unsigned short* orow = Ob + ((size_t)b * M_ + m) * 1024 + h * 64;
#pragma unroll
    for (int nt = 0; nt < 4; nt++) orow[nt * 16 + lr] = f2b(accO[nt][r] * inv);
  }
}

// ------------------------------- GEGLU -------------------------------------
// h: [8192, 4096] bf16 -> g: [8192, 2048] bf16 ; tanh GELU(h1) * h2
__global__ void __launch_bounds__(256) geglu_kernel(
    const unsigned short* __restrict__ hbuf, unsigned short* __restrict__ g) {
  size_t idx = (size_t)blockIdx.x * 256 + threadIdx.x;
  size_t row = idx >> 11, col = idx & 2047;
  const unsigned short* hr = hbuf + row * 4096;
  float x = b2f(hr[col]);
  float y = b2f(hr[col + 2048]);
  float gel = 0.5f * x * (1.0f + tanhf(0.7978845608028654f * (x + 0.044715f * x * x * x)));
  g[idx] = f2b(gel * y);
}

// ---------------------------------------------------------------------------
extern "C" void kernel_launch(void* const* d_in, const int* in_sizes, int n_in,
                              void* d_out, int out_size, void* d_ws, size_t ws_size,
                              hipStream_t stream) {
  (void)in_sizes; (void)n_in; (void)out_size; (void)ws_size;
  const float* hid   = (const float*)d_in[0];
  const int*   amask = (const int*)d_in[1];
  const float* ln1w = (const float*)d_in[2];
  const float* ln1b = (const float*)d_in[3];
  const float* ln2w = (const float*)d_in[4];
  const float* ln2b = (const float*)d_in[5];
  const float* Uq = (const float*)d_in[6];
  const float* Uk = (const float*)d_in[7];
  const float* Uv = (const float*)d_in[8];
  const float* Vq = (const float*)d_in[9];
  const float* Vk = (const float*)d_in[10];
  const float* Vv = (const float*)d_in[11];
  const float* bq = (const float*)d_in[12];
  const float* bk = (const float*)d_in[13];
  const float* bv = (const float*)d_in[14];
  const float* Wo = (const float*)d_in[15];
  const float* Wob = (const float*)d_in[16];
  const float* U1 = (const float*)d_in[17];
  const float* V1 = (const float*)d_in[18];
  const float* b1 = (const float*)d_in[19];
  const float* U2 = (const float*)d_in[20];
  const float* V2 = (const float*)d_in[21];
  const float* b2 = (const float*)d_in[22];
  float* outp = (float*)d_out;

  char* ws = (char*)d_ws;
  // bf16 weight slab
  unsigned short* wbase = (unsigned short*)ws;
  unsigned short* wUq = wbase + 0;
  unsigned short* wUk = wbase + 262144;
  unsigned short* wUv = wbase + 524288;
  unsigned short* wVq = wbase + 786432;
  unsigned short* wVk = wbase + 1048576;
  unsigned short* wVv = wbase + 1310720;
  unsigned short* wWo = wbase + 1572864;   // 1024*1024
  unsigned short* wU1 = wbase + 2621440;   // 1024*512
  unsigned short* wV1 = wbase + 3145728;   // 512*4096
  unsigned short* wU2 = wbase + 5242880;   // 2048*512
  unsigned short* wV2 = wbase + 6291456;   // 512*1024  (ends at 6815744 elems)
  const size_t offA = 13631488;            // 16MB region: xn / xn2
  const size_t offB = offA + 16777216;     // 16MB: Pq,Pk,Pv -> Pf,t
  const size_t offC = offB + 16777216;     // 48MB: Qd,Kd,Vd -> g
  const size_t offD = offC + 50331648;     // 48MB: Qr,Kr,Vr -> h (with E)
  const size_t offE = offD + 50331648;     // 16MB: attn_out bf16
  const size_t offF = offE + 16777216;     // 32MB: x1 f32

  unsigned short* xn    = (unsigned short*)(ws + offA);
  unsigned short* Pq    = (unsigned short*)(ws + offB);
  unsigned short* Pk    = Pq + 2097152;
  unsigned short* Pv    = Pq + 4194304;
  unsigned short* Qd    = (unsigned short*)(ws + offC);
  unsigned short* Kd    = Qd + 8388608;
  unsigned short* Vd    = Qd + 16777216;
  unsigned short* Qr    = (unsigned short*)(ws + offD);
  unsigned short* Kr    = Qr + 8388608;
  unsigned short* Vr    = Qr + 16777216;
  unsigned short* attnb = (unsigned short*)(ws + offE);
  float*          x1    = (float*)(ws + offF);
  unsigned short* xn2   = xn;                                   // reuse A
  unsigned short* Pf    = (unsigned short*)(ws + offB);         // reuse B
  unsigned short* tb    = (unsigned short*)(ws + offB + 8388608);
  unsigned short* hb    = (unsigned short*)(ws + offD);         // reuse D+E (64MB)
  unsigned short* gb    = (unsigned short*)(ws + offC);         // reuse C

  const int TOK = 8192;  // B*M

  // 1) convert weights to bf16
  struct { const float* s; unsigned short* d; int n; } cv[11] = {
    {Uq, wUq, 262144}, {Uk, wUk, 262144}, {Uv, wUv, 262144},
    {Vq, wVq, 262144}, {Vk, wVk, 262144}, {Vv, wVv, 262144},
    {Wo, wWo, 1048576}, {U1, wU1, 524288}, {V1, wV1, 2097152},
    {U2, wU2, 1048576}, {V2, wV2, 524288}};
  for (int i = 0; i < 11; i++)
    cvt_kernel<<<(cv[i].n + 255) / 256, 256, 0, stream>>>(cv[i].s, cv[i].d, cv[i].n);

  // 2) LN1
  ln_kernel<<<TOK, 256, 0, stream>>>(hid, ln1w, ln1b, xn, 1024);

  // 3) rank projections: P* = xn @ U*   [8192 x 256]
  dim3 blk(256);
  gemm_kernel<0,1,0,0><<<dim3(256/64, TOK/128), blk, 0, stream>>>(xn, wUq, nullptr, nullptr, Pq, nullptr, TOK, 256, 1024);
  gemm_kernel<0,1,0,0><<<dim3(256/64, TOK/128), blk, 0, stream>>>(xn, wUk, nullptr, nullptr, Pk, nullptr, TOK, 256, 1024);
  gemm_kernel<0,1,0,0><<<dim3(256/64, TOK/128), blk, 0, stream>>>(xn, wUv, nullptr, nullptr, Pv, nullptr, TOK, 256, 1024);

  // 4) dense Q/K/V = P* @ V* + b*   [8192 x 1024]
  gemm_kernel<0,1,1,0><<<dim3(1024/64, TOK/128), blk, 0, stream>>>(Pq, wVq, bq, nullptr, Qd, nullptr, TOK, 1024, 256);
  gemm_kernel<0,1,1,0><<<dim3(1024/64, TOK/128), blk, 0, stream>>>(Pk, wVk, bk, nullptr, Kd, nullptr, TOK, 1024, 256);
  gemm_kernel<0,1,1,0><<<dim3(1024/64, TOK/128), blk, 0, stream>>>(Pv, wVv, bv, nullptr, Vd, nullptr, TOK, 1024, 256);

  // 5) RoPE (+ head transpose); V transpose only
  rope_kernel<<<4 * 2048 * 16 * 32 / 256, 256, 0, stream>>>(Qd, Qr, 1);
  rope_kernel<<<4 * 2048 * 16 * 32 / 256, 256, 0, stream>>>(Kd, Kr, 1);
  rope_kernel<<<4 * 2048 * 16 * 32 / 256, 256, 0, stream>>>(Vd, Vr, 0);

  // 6) flash attention -> attn_out [B, M, 1024] bf16
  flash_kernel<<<dim3(2048 / 64, 16, 4), 128, 0, stream>>>(Qr, Kr, Vr, amask, attnb);

  // 7) x1 = hidden + attn_out @ Wo^T + Wo_b   (f32)
  gemm_kernel<1,0,1,1><<<dim3(1024/64, TOK/128), blk, 0, stream>>>(attnb, wWo, Wob, hid, nullptr, x1, TOK, 1024, 1024);

  // 8) LN2
  ln_kernel<<<TOK, 256, 0, stream>>>(x1, ln2w, ln2b, xn2, 1024);

  // 9) FFN: P = xn2 @ U1 [8192 x 512]
  gemm_kernel<0,1,0,0><<<dim3(512/64, TOK/128), blk, 0, stream>>>(xn2, wU1, nullptr, nullptr, Pf, nullptr, TOK, 512, 1024);
  // 10) h = P @ V1 + b1 [8192 x 4096]
  gemm_kernel<0,1,1,0><<<dim3(4096/64, TOK/128), blk, 0, stream>>>(Pf, wV1, b1, nullptr, hb, nullptr, TOK, 4096, 512);
  // 11) g = gelu(h1) * h2 [8192 x 2048]
  geglu_kernel<<<TOK * 2048 / 256, 256, 0, stream>>>(hb, gb);
  // 12) t = g @ U2 [8192 x 512]
  gemm_kernel<0,1,0,0><<<dim3(512/64, TOK/128), blk, 0, stream>>>(gb, wU2, nullptr, nullptr, tb, nullptr, TOK, 512, 2048);
  // 13) out = x1 + t @ V2 + b2   (f32 to d_out)
  gemm_kernel<0,0,1,1><<<dim3(1024/64, TOK/128), blk, 0, stream>>>(tb, wV2, b2, x1, nullptr, outp, TOK, 1024, 512);
}
